// AGCN_56684978372722
// MI455X (gfx1250) — matrix operand order repaired and checked by reading, not compiled, not run on hardware
//
#include <hip/hip_runtime.h>
#include <math.h>

#define N_NODES   25000
#define N_EDGES   400000
#define DIM       64
#define EMB       8
#define N_GRAPHS  16
#define N_CLASSES 10
#define HID2      32
#define WAVES     8   // 256-thread blocks = 8 wave32s
#define TPW       4   // tiles (16 edges each) per wave; block covers 512 edges

typedef _Float16 v16h __attribute__((ext_vector_type(16)));
typedef _Float16 v8h  __attribute__((ext_vector_type(8)));
typedef float    v8f  __attribute__((ext_vector_type(8)));

// ---------------- zero kernels ----------------
__global__ void k_zero_out(float* out, int n) {
  int i = blockIdx.x * blockDim.x + threadIdx.x;
  if (i < n) out[i] = 0.f;
}

__global__ void k_zero_layer(float* agg, float* stats, float* pooled) {
  int i = blockIdx.x * blockDim.x + threadIdx.x;
  int stride = gridDim.x * blockDim.x;
  for (int j = i; j < N_NODES * DIM; j += stride) agg[j] = 0.f;
  if (i < 2 * DIM) stats[i] = 0.f;
  if (i < N_GRAPHS * DIM) pooled[i] = 0.f;
}

// ---------------- f32 -> f16 feature mirror (layer-0 input) ----------------
__global__ void k_h2half(const float* __restrict__ x, _Float16* __restrict__ y, int n) {
  int i = blockIdx.x * blockDim.x + threadIdx.x;
  int stride = gridDim.x * blockDim.x;
  for (int j = i; j < n; j += stride) y[j] = (_Float16)x[j];
}

// ---------------- W (8,64,64) f32 -> Wt (8,64,64) f16, transposed [i][n][j] ----------------
__global__ void k_wcvt(const float* __restrict__ W, _Float16* __restrict__ Wt) {
  int f = blockIdx.x * blockDim.x + threadIdx.x;
  if (f >= EMB * DIM * DIM) return;
  int i = f >> 12;
  int j = (f >> 6) & 63;   // K (input) dim
  int n = f & 63;          // N (output) dim
  Wt[(i << 12) + (n << 6) + j] = (_Float16)W[f];
}

// ---------------- edge bilinear via WMMA + atomic scatter ----------------
// Per wave: 4 tiles of 16 edges; C(16x64,f32) = sum_i (d_i * h_src)(16x64,f16) @ W_i(64x64,f16)
// Full transposed W staged in dynamic LDS once per block; B frags served by ds_load_b128.
// h gathered from pre-converted f16 mirror (half the HBM bytes of f32).
__global__ __launch_bounds__(32 * WAVES)
void k_edge(const int* __restrict__ src, const int* __restrict__ dst,
            const float* __restrict__ cemb, const _Float16* __restrict__ h16,
            const _Float16* __restrict__ Wt, float* __restrict__ agg) {
  __shared__ __align__(16) _Float16 hsSh[WAVES][16][DIM];   // f16 h[src] tiles
  __shared__ float dSh[WAVES][16][EMB];                     // normalized directions
  __shared__ int   dstSh[WAVES][16];
  extern __shared__ __align__(16) char dynLds[];            // 64 KB: transposed W in f16
  _Float16* WtLds = (_Float16*)dynLds;

  const int lane = threadIdx.x & 31;
  const int wave = threadIdx.x >> 5;
  const int m  = lane & 15;   // matrix row this lane serves
  const int hi = lane >> 4;   // which half of the 64 cols to stage

  // --- stage full Wt into LDS (4096 uint4, 16 per thread) ---
  {
    const uint4* gsrc = (const uint4*)Wt;
    uint4* ldst = (uint4*)WtLds;
    for (int q = threadIdx.x; q < (EMB * DIM * DIM) / 8; q += blockDim.x) ldst[q] = gsrc[q];
  }

  const int blockBase = blockIdx.x * (WAVES * TPW * 16);
  const int kA   = (lane & 16) ? 8 : 0;   // A-layout K base per lane-half (ISA 7.12.2)
  const int kB   = (lane & 16) ? 8 : 0;   // B-layout K base per lane-half (same as A)
  const int ncol = lane & 15;             // B/C column within n-tile

#pragma unroll 1
  for (int t = 0; t < TPW; ++t) {
    const int tileBase = blockBase + (wave * TPW + t) * 16;

    __syncthreads();  // covers Wt staging (t==0) and hsSh WAR across iterations

    // --- stage direction vectors + dst indices (lanes 0..15: one edge each) ---
    if (lane < 16) {
      int e = tileBase + lane;
      bool valid = e < N_EDGES;
      int s = valid ? src[e] : 0;
      int d = valid ? dst[e] : 0;
      dstSh[wave][lane] = valid ? d : -1;
      const float* cs = cemb + (size_t)s * EMB;
      const float* ct = cemb + (size_t)d * EMB;
      float dv[EMB]; float nrm = 0.f;
#pragma unroll
      for (int i = 0; i < EMB; ++i) { float x = cs[i] - ct[i]; dv[i] = x; nrm += x * x; }
      float sc = valid ? 1.f / fmaxf(sqrtf(nrm), 1e-12f) : 0.f;
#pragma unroll
      for (int i = 0; i < EMB; ++i) dSh[wave][lane][i] = dv[i] * sc;
    }

    // --- stage h16[src] rows (all 32 lanes: lane -> row m, 64B half 'hi') ---
    {
      int e = tileBase + m;
      int s = (e < N_EDGES) ? src[e] : 0;
      const uint4* hp = (const uint4*)(h16 + (size_t)s * DIM + hi * 32);
      uint4* lp = (uint4*)&hsSh[wave][m][hi * 32];
#pragma unroll
      for (int q = 0; q < 4; ++q) lp[q] = hp[q];
    }
    __syncthreads();

    // --- WMMA accumulation: 8 i-chunks x 2 k-steps x 4 n-tiles = 64 wmma ---
    v8f acc[4] = {};  // 16x64 f32 accumulator (4 n-tiles)
#pragma unroll
    for (int i = 0; i < EMB; ++i) {
      _Float16 dsc = (_Float16)dSh[wave][m][i];
#pragma unroll
      for (int ks = 0; ks < 2; ++ks) {
        const int jb = ks * 32 + kA;
        v8h a0 = *(const v8h*)&hsSh[wave][m][jb];        // K = jb .. jb+7
        v8h a1 = *(const v8h*)&hsSh[wave][m][jb + 16];   // K = jb+16 .. jb+23
        a0 = a0 * dsc;                                   // row-scale by d[e,i]
        a1 = a1 * dsc;
        v16h A;
#pragma unroll
        for (int q = 0; q < 8; ++q) { A[q] = a0[q]; A[8 + q] = a1[q]; }
#pragma unroll
        for (int nt = 0; nt < 4; ++nt) {
          const _Float16* wp = WtLds + (i << 12) + ((nt * 16 + ncol) << 6) + (ks * 32 + kB);
          v8h b0 = *(const v8h*)wp;        // K .. K+7 of column n
          v8h b1 = *(const v8h*)(wp + 16); // K+16 .. K+23
          v16h B;
#pragma unroll
          for (int q = 0; q < 8; ++q) { B[q] = b0[q]; B[8 + q] = b1[q]; }
          acc[nt] = __builtin_amdgcn_wmma_f32_16x16x32_f16(
              false, A, false, B, (short)0, acc[nt], false, false);
        }
      }
    }

    // --- scatter rows to agg[dst] per C layout (VGPR r: M=r(+8), lane: N) ---
#pragma unroll
    for (int r = 0; r < 8; ++r) {
      int mrow = r + ((lane & 16) ? 8 : 0);
      int dnode = dstSh[wave][mrow];
      if (dnode >= 0) {
        float* ap = agg + (size_t)dnode * DIM + ncol;
#pragma unroll
        for (int nt = 0; nt < 4; ++nt) atomicAdd(ap + nt * 16, acc[nt][r]);
      }
    }
  }
}

// ---------------- BN column stats over relu(agg) ----------------
__global__ void k_stats(const float* __restrict__ agg, float* __restrict__ stats) {
  int tid = blockIdx.x * blockDim.x + threadIdx.x;
  int c = tid & 63;
  int r = tid >> 6;
  int rs = (gridDim.x * blockDim.x) >> 6;
  float s = 0.f, s2 = 0.f;
  for (; r < N_NODES; r += rs) {
    float x = fmaxf(agg[(size_t)r * DIM + c], 0.f);
    s += x; s2 += x * x;
  }
  atomicAdd(&stats[c], s);
  atomicAdd(&stats[DIM + c], s2);
}

// ---------------- normalize + shortcut + graph pooling + f16 mirror ----------------
__global__ void k_bnorm_pool(const float* __restrict__ agg, const float* __restrict__ stats,
                             const float* __restrict__ hprev, const float* __restrict__ gamma,
                             const float* __restrict__ beta, const int* __restrict__ gids,
                             float* __restrict__ hout, _Float16* __restrict__ hout16,
                             float* __restrict__ pooled, int shortcut) {
  int i = blockIdx.x * blockDim.x + threadIdx.x;
  int stride = gridDim.x * blockDim.x;
  const float invN = 1.f / (float)N_NODES;
  for (int j = i; j < N_NODES * DIM; j += stride) {
    int c = j & 63; int r = j >> 6;
    float mu  = stats[c] * invN;
    float var = stats[DIM + c] * invN - mu * mu;
    float inv = rsqrtf(var + 1e-5f);
    float x = fmaxf(agg[j], 0.f);
    float o = gamma[c] * (x - mu) * inv + beta[c];
    if (shortcut) o += hprev[j];
    hout[j] = o;
    hout16[j] = (_Float16)o;
    atomicAdd(&pooled[(size_t)gids[r] * DIM + c], o);
  }
}

// ---------------- tiny classifier on pooled (16x64), accumulates into logits ----------------
__global__ void k_classifier(const float* __restrict__ pooled,
                             const float* __restrict__ cW1, const float* __restrict__ cb1,
                             const float* __restrict__ cg1, const float* __restrict__ cbt1,
                             const float* __restrict__ cW2, const float* __restrict__ cb2,
                             float* __restrict__ out) {
  __shared__ float zsh[N_GRAPHS][HID2];
  __shared__ float mus[HID2], ivs[HID2];
  __shared__ float h1[N_GRAPHS][HID2];
  int t = threadIdx.x;
  if (t < N_GRAPHS * HID2) {
    int r = t / HID2, c = t % HID2;
    float s = cb1[c];
    for (int j = 0; j < DIM; ++j) s += pooled[r * DIM + j] * cW1[j * HID2 + c];
    zsh[r][c] = s;
  }
  __syncthreads();
  if (t < HID2) {
    float s = 0.f, s2 = 0.f;
    for (int r = 0; r < N_GRAPHS; ++r) { float x = zsh[r][t]; s += x; s2 += x * x; }
    float mu = s / (float)N_GRAPHS;
    mus[t] = mu;
    ivs[t] = rsqrtf(s2 / (float)N_GRAPHS - mu * mu + 1e-5f);
  }
  __syncthreads();
  if (t < N_GRAPHS * HID2) {
    int r = t / HID2, c = t % HID2;
    float v = cg1[c] * (zsh[r][c] - mus[c]) * ivs[c] + cbt1[c];
    h1[r][c] = fmaxf(v, 0.f);
  }
  __syncthreads();
  if (t < N_GRAPHS * N_CLASSES) {
    int r = t / N_CLASSES, k = t % N_CLASSES;
    float s = cb2[k];
    for (int c = 0; c < HID2; ++c) s += h1[r][c] * cW2[c * N_CLASSES + k];
    out[r * N_CLASSES + k] += s;
  }
}

// ---------------- host orchestration ----------------
extern "C" void kernel_launch(void* const* d_in, const int* in_sizes, int n_in,
                              void* d_out, int out_size, void* d_ws, size_t ws_size,
                              hipStream_t stream) {
  const float* feature = (const float*)d_in[0];
  const float* cemb    = (const float*)d_in[1];
  const int*   src     = (const int*)d_in[2];
  const int*   dst     = (const int*)d_in[3];
  const int*   gids    = (const int*)d_in[4];
  const float* Ws[3]   = {(const float*)d_in[5], (const float*)d_in[6], (const float*)d_in[7]};
  const float* gm[3]   = {(const float*)d_in[8], (const float*)d_in[10], (const float*)d_in[12]};
  const float* bt[3]   = {(const float*)d_in[9], (const float*)d_in[11], (const float*)d_in[13]};
  const float* cW1  = (const float*)d_in[14];
  const float* cb1  = (const float*)d_in[15];
  const float* cg1  = (const float*)d_in[16];
  const float* cbt1 = (const float*)d_in[17];
  const float* cW2  = (const float*)d_in[18];
  const float* cb2  = (const float*)d_in[19];
  float* out = (float*)d_out;

  char* p = (char*)d_ws;
  auto carve = [&](size_t bytes) -> char* {
    char* q = p; p += (bytes + 255) & ~(size_t)255; return q;
  };
  float*    agg    = (float*)carve(sizeof(float) * N_NODES * DIM);
  float*    hA     = (float*)carve(sizeof(float) * N_NODES * DIM);
  float*    hB     = (float*)carve(sizeof(float) * N_NODES * DIM);
  _Float16* feat16 = (_Float16*)carve(sizeof(_Float16) * N_NODES * DIM);
  _Float16* h16A   = (_Float16*)carve(sizeof(_Float16) * N_NODES * DIM);
  _Float16* h16B   = (_Float16*)carve(sizeof(_Float16) * N_NODES * DIM);
  _Float16* Wt     = (_Float16*)carve(sizeof(_Float16) * EMB * DIM * DIM);
  float*    stats  = (float*)carve(sizeof(float) * 2 * DIM);
  float*    pooled = (float*)carve(sizeof(float) * N_GRAPHS * DIM);

  k_zero_out<<<1, 256, 0, stream>>>(out, N_GRAPHS * N_CLASSES);
  k_h2half<<<512, 256, 0, stream>>>(feature, feat16, N_NODES * DIM);

  const float*    hin   = feature;
  const _Float16* hin16 = feat16;
  float*    houts[3]   = {hA, hB, hA};
  _Float16* houts16[3] = {h16A, h16B, h16A};
  const int edgesPerBlock = WAVES * TPW * 16;  // 512
  const int edgeBlocks = (N_EDGES + edgesPerBlock - 1) / edgesPerBlock;  // 782
  const size_t dynShmem = sizeof(_Float16) * EMB * DIM * DIM;            // 64 KB
  for (int L = 0; L < 3; ++L) {
    k_zero_layer<<<1024, 256, 0, stream>>>(agg, stats, pooled);
    k_wcvt<<<(EMB * DIM * DIM + 255) / 256, 256, 0, stream>>>(Ws[L], Wt);
    k_edge<<<edgeBlocks, 32 * WAVES, dynShmem, stream>>>(src, dst, cemb, hin16, Wt, agg);
    k_stats<<<128, 256, 0, stream>>>(agg, stats);
    k_bnorm_pool<<<1024, 256, 0, stream>>>(agg, stats, hin, gm[L], bt[L], gids,
                                           houts[L], houts16[L], pooled, (L > 0) ? 1 : 0);
    k_classifier<<<1, 512, 0, stream>>>(pooled, cW1, cb1, cg1, cbt1, cW2, cb2, out);
    hin = houts[L];
    hin16 = houts16[L];
  }
  (void)in_sizes; (void)n_in; (void)out_size; (void)ws_size;
}
